// PureSlidingWindowAttention_10582799417770
// MI455X (gfx1250) — matrix operand order repaired and checked by reading, not compiled, not run on hardware
//
#include <hip/hip_runtime.h>

typedef __bf16 bf16_t;
typedef __attribute__((ext_vector_type(16))) __bf16 v16bf;
typedef __attribute__((ext_vector_type(8)))  __bf16 v8bf;
typedef __attribute__((ext_vector_type(8)))  float  v8f;

#define B_DIM 4
#define T_DIM 2048
#define D_DIM 256
#define C_DIM 64
#define NHEAD 4
#define BT    (B_DIM * T_DIM)    // 8192 rows
#define NQKV  384                // nh*c + c + c
#define KDIM  256                // K for both WMMA GEMMs

// ---------------------------------------------------------------------------
// fp32 -> bf16 elementwise conversion
// ---------------------------------------------------------------------------
__global__ void cvt_f32_to_bf16(const float* __restrict__ in,
                                bf16_t* __restrict__ out, int n) {
  int i = blockIdx.x * blockDim.x + threadIdx.x;
  if (i < n) out[i] = (bf16_t)in[i];
}

// ---------------------------------------------------------------------------
// Fuse:  W_final[o][i] = sum_j Wout[o][j] * blockdiag(Wg0,Wg1)[j][i]
//        b_final[o]    = bout[o] + sum_j Wout[o][j] * bg[j]
// Launch: <<<256, 256>>>  (o = blockIdx.x, i = threadIdx.x)
// ---------------------------------------------------------------------------
__global__ void fuse_out_weights(const float* __restrict__ Wg0,
                                 const float* __restrict__ Wg1,
                                 const float* __restrict__ Wout,
                                 const float* __restrict__ bg0,
                                 const float* __restrict__ bg1,
                                 const float* __restrict__ bout,
                                 bf16_t* __restrict__ Wf,
                                 float* __restrict__ bfin) {
  int o = blockIdx.x;
  int i = threadIdx.x;
  float s = 0.f;
  if (i < 128) {
    for (int j = 0; j < 64; ++j) s += Wout[o * 128 + j] * Wg0[j * 128 + i];
  } else {
    for (int j = 0; j < 64; ++j) s += Wout[o * 128 + 64 + j] * Wg1[j * 128 + (i - 128)];
  }
  Wf[o * 256 + i] = (bf16_t)s;
  if (i == 0) {
    float b = bout[o];
    for (int j = 0; j < 64; ++j)
      b += Wout[o * 128 + j] * bg0[j] + Wout[o * 128 + 64 + j] * bg1[j];
    bfin[o] = b;
  }
}

// ---------------------------------------------------------------------------
// WMMA bf16 GEMM, 2x2 register-blocked:
//   C[M,N] = A[M,K] (row-major) @ B[N,K]^T (+ bias[N])
// One wave computes a 32x32 macro-tile (four 16x16 WMMA tiles): per k-step
// 2 A-frags + 2 B-frags feed 4 WMMAs -> 2 b128 loads per WMMA (2x the
// arithmetic intensity of 1-tile-per-wave; halves L2 traffic).
// Fragment layouts per CDNA5 ISA 7.12.2:
//   A 16x32: lane L (r=L&15, kb=(L>>4)*8)   holds K=[kb..kb+7], K=[kb+16..kb+23]
//   B 32x16: lane L (n=L&15, kb=(L>>4)*8)   holds K=[kb..kb+7], K=[kb+16..kb+23] of column n
//   D 16x16: VGPR j -> M = j + ((lane>=16)?8:0), N = lane&15
// ---------------------------------------------------------------------------
union frag_u { v16bf v; v8bf h[2]; };

__device__ __forceinline__ v16bf load_frag_A(const bf16_t* __restrict__ base,
                                             int lda, int lane) {
  int r  = lane & 15;
  int kb = (lane >> 4) << 3;             // 0 or 8
  const bf16_t* p = base + (size_t)r * lda + kb;
  frag_u u;
  u.h[0] = *(const v8bf*)(p);            // K = kb .. kb+7
  u.h[1] = *(const v8bf*)(p + 16);       // K = kb+16 .. kb+23
  return u.v;
}

__device__ __forceinline__ v16bf load_frag_B(const bf16_t* __restrict__ base,
                                             int ldb, int lane) {
  int n  = lane & 15;
  int kb = (lane >> 4) << 3;             // 0 or 8
  const bf16_t* p = base + (size_t)n * ldb + kb;
  frag_u u;
  u.h[0] = *(const v8bf*)(p);            // K = kb .. kb+7
  u.h[1] = *(const v8bf*)(p + 16);       // K = kb+16 .. kb+23
  return u.v;
}

__device__ __forceinline__ void store_tile(float* __restrict__ C, int N,
                                           int m0, int n0, float bv,
                                           const v8f& acc, int lane) {
  int n  = lane & 15;
  int mh = (lane >> 4) << 3;             // 0 or 8
#pragma unroll
  for (int j = 0; j < 8; ++j) {
    int row = m0 + mh + j;
    C[(size_t)row * N + n0 + n] = acc[j] + bv;
  }
}

__global__ void wmma_gemm_bf16_2x2(const bf16_t* __restrict__ A,
                                   const bf16_t* __restrict__ B,
                                   float* __restrict__ C,
                                   const float* __restrict__ bias,
                                   int M, int N) {
  int waveId = blockIdx.x * (blockDim.x >> 5) + (threadIdx.x >> 5);
  int lane   = threadIdx.x & 31;
  int tilesN = N >> 5;                   // 32-wide macro tiles
  int tm = waveId / tilesN;
  int tn = waveId - tm * tilesN;
  if (tm >= (M >> 5)) return;            // wave-uniform guard (EXEC stays full)
  int m0 = tm << 5, n0 = tn << 5;

  v8f acc00 = {0.f, 0.f, 0.f, 0.f, 0.f, 0.f, 0.f, 0.f};
  v8f acc01 = acc00, acc10 = acc00, acc11 = acc00;

#pragma unroll
  for (int k0 = 0; k0 < KDIM; k0 += 32) {
    v16bf a0 = load_frag_A(A + (size_t)m0 * KDIM + k0,        KDIM, lane);
    v16bf a1 = load_frag_A(A + (size_t)(m0 + 16) * KDIM + k0, KDIM, lane);
    v16bf b0 = load_frag_B(B + (size_t)n0 * KDIM + k0,        KDIM, lane);
    v16bf b1 = load_frag_B(B + (size_t)(n0 + 16) * KDIM + k0, KDIM, lane);
    acc00 = __builtin_amdgcn_wmma_f32_16x16x32_bf16(false, a0, false, b0,
                                                    (short)0, acc00, false, false);
    acc01 = __builtin_amdgcn_wmma_f32_16x16x32_bf16(false, a0, false, b1,
                                                    (short)0, acc01, false, false);
    acc10 = __builtin_amdgcn_wmma_f32_16x16x32_bf16(false, a1, false, b0,
                                                    (short)0, acc10, false, false);
    acc11 = __builtin_amdgcn_wmma_f32_16x16x32_bf16(false, a1, false, b1,
                                                    (short)0, acc11, false, false);
  }

  int n   = lane & 15;
  float bv0 = bias ? bias[n0 + n]      : 0.f;
  float bv1 = bias ? bias[n0 + 16 + n] : 0.f;
  store_tile(C, N, m0,      n0,      bv0, acc00, lane);
  store_tile(C, N, m0,      n0 + 16, bv1, acc01, lane);
  store_tile(C, N, m0 + 16, n0,      bv0, acc10, lane);
  store_tile(C, N, m0 + 16, n0 + 16, bv1, acc11, lane);
}

// ---------------------------------------------------------------------------
// RoPE on last 16 dims of a 64-dim vector (thetas = 10000^(-k/8), constants)
// ---------------------------------------------------------------------------
__device__ __forceinline__ void rope16(float* x, float pos, float sgn) {
  const float thetas[8] = {1.f, 0.316227766016838f, 0.1f, 0.0316227766016838f,
                           0.01f, 0.00316227766016838f, 0.001f, 0.000316227766016838f};
#pragma unroll
  for (int k = 0; k < 8; ++k) {
    float f  = pos * thetas[k];
    float cs = cosf(f);
    float sn = sinf(f) * sgn;
    float x1 = x[48 + 2 * k];
    float x2 = x[48 + 2 * k + 1];
    x[48 + 2 * k]     = x1 * cs - x2 * sn;
    x[48 + 2 * k + 1] = x1 * sn + x2 * cs;
  }
}

// ---------------------------------------------------------------------------
// In-place rmsnorm + RoPE on the projected Q(4 heads)/K/V rows.
// One thread per 64-dim vector: gid = vec*BT + row, vec in [0,6).
// ---------------------------------------------------------------------------
__global__ void qkv_post(float* __restrict__ qkv,
                         const float* __restrict__ wq,
                         const float* __restrict__ wk,
                         const float* __restrict__ wv) {
  int gid = blockIdx.x * blockDim.x + threadIdx.x;
  if (gid >= BT * 6) return;
  int row = gid % BT;
  int vec = gid / BT;                    // 0..3 = Q heads, 4 = K, 5 = V
  int t   = row % T_DIM;
  int off = (vec < 4) ? vec * 64 : (vec == 4 ? 256 : 320);
  const float* w = (vec < 4) ? (wq + vec * 64) : (vec == 4 ? wk : wv);
  float* p = qkv + (size_t)row * NQKV + off;

  float x[64];
  float ss = 0.f;
#pragma unroll
  for (int i = 0; i < 64; ++i) { x[i] = p[i]; ss += x[i] * x[i]; }
  float inv = rsqrtf(ss * (1.f / 64.f) + 1e-8f);
#pragma unroll
  for (int i = 0; i < 64; ++i) x[i] = x[i] * inv * w[i];
  rope16(x, (float)t, 1.f);
#pragma unroll
  for (int i = 0; i < 64; ++i) p[i] = x[i];
}

// ---------------------------------------------------------------------------
// Sliding-window (16) causal attention. One thread per (b, h, t).
// Reads post-processed qkv, writes negate-RoPE'd O as bf16 [BT, 256].
// ---------------------------------------------------------------------------
__global__ void attn_window(const float* __restrict__ qkv,
                            bf16_t* __restrict__ Ob) {
  int gid = blockIdx.x * blockDim.x + threadIdx.x;
  if (gid >= B_DIM * NHEAD * T_DIM) return;
  int t = gid % T_DIM;
  int h = (gid / T_DIM) & 3;
  int b = gid / (NHEAD * T_DIM);
  size_t base = (size_t)(b * T_DIM) * NQKV;

  const float* qp = qkv + base + (size_t)t * NQKV + h * 64;
  float q[64];
#pragma unroll
  for (int i = 0; i < 64; ++i) q[i] = qp[i];

  float w[16];
  float m = -3.0e38f;
#pragma unroll
  for (int si = 0; si < 16; ++si) {
    int s = t - 15 + si;
    float dot = -3.0e38f;
    if (s >= 0) {
      const float* kp = qkv + base + (size_t)s * NQKV + 256;
      float d = 0.f;
#pragma unroll
      for (int i = 0; i < 64; ++i) d += q[i] * kp[i];
      dot = d * 0.125f;                  // 1/sqrt(64)
    }
    w[si] = dot;
    m = fmaxf(m, dot);
  }

  float den = 0.f;
#pragma unroll
  for (int si = 0; si < 16; ++si) {
    float e = (w[si] > -1.0e37f) ? __expf(w[si] - m) : 0.f;
    w[si] = e;
    den += e;
  }
  float inv_den = 1.f / den;

  float o[64];
#pragma unroll
  for (int i = 0; i < 64; ++i) o[i] = 0.f;
#pragma unroll
  for (int si = 0; si < 16; ++si) {
    int s = t - 15 + si;
    if (s >= 0) {
      float wt = w[si] * inv_den;
      const float* vp = qkv + base + (size_t)s * NQKV + 320;
#pragma unroll
      for (int i = 0; i < 64; ++i) o[i] += wt * vp[i];
    }
  }

  rope16(o, (float)t, -1.f);             // negate=True

  bf16_t* op = Ob + ((size_t)(b * T_DIM) + t) * 256 + h * 64;
#pragma unroll
  for (int i = 0; i < 64; ++i) op[i] = (bf16_t)o[i];
}

// ---------------------------------------------------------------------------
// Launch
// ---------------------------------------------------------------------------
extern "C" void kernel_launch(void* const* d_in, const int* in_sizes, int n_in,
                              void* d_out, int out_size, void* d_ws, size_t ws_size,
                              hipStream_t stream) {
  const float* H    = (const float*)d_in[0];   // 8192 x 256
  const float* W_Q  = (const float*)d_in[1];   // 256 x 256
  const float* W_K  = (const float*)d_in[2];   // 64 x 256
  const float* W_V  = (const float*)d_in[3];   // 64 x 256
  const float* wq   = (const float*)d_in[4];   // 4 x 64
  const float* wk   = (const float*)d_in[5];   // 64
  const float* wv   = (const float*)d_in[6];   // 64
  const float* Wg0  = (const float*)d_in[7];   // 64 x 128
  const float* bg0  = (const float*)d_in[8];   // 64
  const float* Wg1  = (const float*)d_in[9];   // 64 x 128
  const float* bg1  = (const float*)d_in[10];  // 64
  const float* Wout = (const float*)d_in[11];  // 256 x 128
  const float* bout = (const float*)d_in[12];  // 256
  float* out = (float*)d_out;                  // 8192 x 256 fp32

  // Workspace layout (all offsets 256B aligned); total ~21.3 MB (L2-resident).
  char* ws = (char*)d_ws;
  bf16_t* Hb   = (bf16_t*)(ws);                 //  4,194,304 B
  bf16_t* Wqkv = (bf16_t*)(ws + 4194304);       //    196,608 B (384x256)
  float*  qkv  = (float*) (ws + 4390912);       // 12,582,912 B (8192x384)
  bf16_t* Ob   = (bf16_t*)(ws + 16973824);      //  4,194,304 B (8192x256)
  bf16_t* Wf   = (bf16_t*)(ws + 21168128);      //    131,072 B (256x256)
  float*  bfin = (float*) (ws + 21299200);      //      1,024 B

  // 1) bf16 conversions
  cvt_f32_to_bf16<<<(BT * D_DIM) / 256, 256, 0, stream>>>(H, Hb, BT * D_DIM);
  cvt_f32_to_bf16<<<256, 256, 0, stream>>>(W_Q, Wqkv,          256 * 256);
  cvt_f32_to_bf16<<<64,  256, 0, stream>>>(W_K, Wqkv + 65536,  64 * 256);
  cvt_f32_to_bf16<<<64,  256, 0, stream>>>(W_V, Wqkv + 81920,  64 * 256);

  // 2) fuse the two output projections into one 256x256 weight + bias
  fuse_out_weights<<<256, 256, 0, stream>>>(Wg0, Wg1, Wout, bg0, bg1, bout, Wf, bfin);

  // 3) QKV projection: [8192,256] @ [384,256]^T -> fp32 [8192,384]
  //    macro tiles = (8192/32) x (384/32) = 256 x 12 = 3072 waves; 8 waves/block
  wmma_gemm_bf16_2x2<<<3072 / 8, 256, 0, stream>>>(Hb, Wqkv, qkv, nullptr,
                                                   BT, NQKV);

  // 4) rmsnorm + RoPE in place on Q/K/V
  qkv_post<<<(BT * 6) / 256, 256, 0, stream>>>(qkv, wq, wk, wv);

  // 5) sliding-window attention + negate-RoPE -> bf16 O
  attn_window<<<(B_DIM * NHEAD * T_DIM) / 256, 256, 0, stream>>>(qkv, Ob);

  // 6) fused output projection: [8192,256] @ [256,256]^T + bias -> d_out
  //    macro tiles = 256 x 8 = 2048 waves
  wmma_gemm_bf16_2x2<<<2048 / 8, 256, 0, stream>>>(Ob, Wf, out, bfin,
                                                   BT, D_DIM);
}
